// SpinConvBlock_31516470018580
// MI455X (gfx1250) — matrix operand order repaired and checked by reading, not compiled, not run on hardware
//
#include <hip/hip_runtime.h>
#include <stdint.h>

// ---------------- problem constants (match reference) ----------------
#define LATD     16
#define LONGD    12
#define C_IN     128
#define C_MID    128
#define GROUPSD  16
#define OCPG     8                    // out channels per group (also GN group size)
#define ICPG     128                  // conv in-channels per group = 8 c * 16 lat
#define SPATIAL  (LATD * LONGD)       // 192 positions per target
#define K_TOT    (8 * SPATIAL)        // 1536 contraction length per group
#define KCH      (K_TOT / 32)         // 48 WMMA K-chunks (bf16 k=32)
#define NCOL     (OCPG * LONGD)       // 96 output columns = (oc, p)
#define NT       (NCOL / 16)          // 6 WMMA column tiles
#define EPSF     1e-5f

// ---------------- CDNA5 WMMA vector types ----------------
typedef __attribute__((ext_vector_type(16))) __bf16 v16bf;
typedef __attribute__((ext_vector_type(8)))  float  v8f;

union BFrag {
    uint4 u[2];   // 32 bytes
    v16bf v;
};

__device__ __forceinline__ unsigned short f2bf(float f) {
    union { float f; uint32_t u; } cv; cv.f = f;
    uint32_t u = cv.u;
    uint32_t r = u + 0x7FFFu + ((u >> 16) & 1u);   // round-to-nearest-even
    return (unsigned short)(r >> 16);
}

// ---------------- kernel 1: zero the (bf16) scatter buffer ----------------
__global__ void zero_ws_kernel(uint4* __restrict__ p, long n16) {
    long i = (long)blockIdx.x * blockDim.x + threadIdx.x;
    long stride = (long)gridDim.x * blockDim.x;
    uint4 z; z.x = 0u; z.y = 0u; z.z = 0u; z.w = 0u;
    for (; i < n16; i += stride) p[i] = z;
}

// ---------------- kernel 2: expand circular conv weights into WMMA B-fragments ----------------
// GEMM per group:  Y[n,(oc,p)] = sum_K A[n,K] * B[K,(oc,p)]
//   K = s*8 + c  (s = lat*12 + j in [0,192), c = original channel in group, [0,8))
//   B[K,(p*8+oc)] = w[oc, c*16+lat, (j - p + 6) mod 12]
// Stored pre-swizzled in WMMA B-fragment lane order:
//   bexp[((((g*KCH + kc)*NT + t)*32 + lane)*16 + q)]
//   lane = col-in-tile (+16 repeat); lanes 0-15 hold K_local 0..15, lanes 16-31 hold 16..31.
__global__ void build_bexp_kernel(const float* __restrict__ w, unsigned short* __restrict__ bexp) {
    int idx = blockIdx.x * blockDim.x + threadIdx.x;
    const int total = GROUPSD * KCH * NT * 32 * 16;
    if (idx >= total) return;
    int q    = idx & 15;
    int lane = (idx >> 4) & 31;
    int rem  = idx >> 9;                 // g*KCH*NT + kc*NT + t
    int t    = rem % NT;
    int kc   = (rem / NT) % KCH;
    int g    = rem / (NT * KCH);

    int klocal = q + ((lane >> 4) << 4); // 0..31
    int col    = (lane & 15) + 16 * t;   // 0..95
    int K      = kc * 32 + klocal;       // 0..1535

    int s   = K >> 3;                    // 0..191 spatial position
    int c   = K & 7;                     // 0..7   original channel within group
    int lat = s / LONGD;
    int j   = s % LONGD;
    int p   = col >> 3;                  // 0..11
    int oc  = col & 7;                   // 0..7
    int k   = (j - p + 18) % 12;         // (j - p + 6) mod 12
    int i   = c * LATD + lat;            // conv in-channel index within group

    float val = w[(((size_t)(g * OCPG + oc) * ICPG) + i) * LONGD + k];
    bexp[idx] = f2bf(val);
}

// ---------------- kernel 3: weighted scatter-add (4 projections, packed-bf16 atomics) ----------------
// one wave (32 lanes) per edge; 4 channels per lane -> 2 x global_atomic_pk_add_bf16.
// Average contributions per destination row ~= 4E/(N*192) ~= 1.0, so bf16
// accumulation adds no more rounding than the bf16 conversion the conv does anyway.
__global__ void scatter_add_kernel(const float* __restrict__ x,
                                   const int* __restrict__ pidx,
                                   const float* __restrict__ pdelta,
                                   const int* __restrict__ psrc,
                                   unsigned short* __restrict__ xproj, int E) {
    long gid = (long)blockIdx.x * blockDim.x + threadIdx.x;
    int  e   = (int)(gid >> 5);
    if (e >= E) return;
    int lane = (int)(gid & 31);

    int src = psrc[e];
    const float4 v = *(const float4*)(x + (size_t)src * C_IN + lane * 4);

#pragma unroll
    for (int i = 0; i < 4; ++i) {
        int   di = pidx[(size_t)i * E + e];
        float d  = pdelta[(size_t)i * E + e];
        unsigned short* dst = xproj + (size_t)di * C_IN + lane * 4;
        uint32_t p0 = (uint32_t)f2bf(v.x * d) | ((uint32_t)f2bf(v.y * d) << 16);
        uint32_t p1 = (uint32_t)f2bf(v.z * d) | ((uint32_t)f2bf(v.w * d) << 16);
        unsigned long long a0 = (unsigned long long)(size_t)dst;
        unsigned long long a1 = a0 + 4ull;
        asm volatile("global_atomic_pk_add_bf16 %0, %1, off"
                     :: "v"(a0), "v"(p0) : "memory");
        asm volatile("global_atomic_pk_add_bf16 %0, %1, off"
                     :: "v"(a1), "v"(p1) : "memory");
    }
}

// ---------------- kernel 4: fused grouped conv (WMMA) + SiLU + avg-pool + GroupNorm ----------------
// block = 64 threads (2 waves); block handles 16 n-rows x 1 group.
// GEMM: M=16 (n), N=96 (oc,p), K=1536; bf16 WMMA, f32 accumulate.
// A tile staged to LDS with async global->LDS b128 copies (pure byte copy since
// K = s*8 + c makes the LDS layout the concatenation of the global 16B segments).
__global__ __launch_bounds__(64) void conv_fused_kernel(
    const unsigned short* __restrict__ xproj, const unsigned short* __restrict__ bexp,
    const float* __restrict__ conv_b, const float* __restrict__ gn_w,
    const float* __restrict__ gn_b, float* __restrict__ out) {

    const int g   = blockIdx.x;            // group (fast dim -> L2 reuse of xproj lines)
    const int n0  = blockIdx.y * 16;
    const int tid = threadIdx.x;           // 0..63
    const int lane = tid & 31;
    const int wave = tid >> 5;

    __shared__ __align__(16) unsigned short As[16 * K_TOT];  // 48 KB bf16 A-tile
    __shared__ float Ls[16][NCOL];                            // conv outputs (pre-SiLU)
    __shared__ float Pm[16][OCPG];                            // pooled means

    // ---- stage A tile: async global->LDS, 16B segment per (n,s) ----
    {
        const unsigned lds0 = (unsigned)(size_t)(&As[0]);
        for (int idx = tid; idx < 16 * SPATIAL; idx += 64) {
            int nl = idx / SPATIAL;
            int s  = idx - nl * SPATIAL;
            unsigned ldst = lds0 + (unsigned)(nl * K_TOT + s * 8) * 2u;  // bytes
            const unsigned short* src =
                xproj + ((size_t)(n0 + nl) * SPATIAL + s) * C_IN + g * 8;
            unsigned long long ga = (unsigned long long)(size_t)src;
            asm volatile("global_load_async_to_lds_b128 %0, %1, off"
                         :: "v"(ldst), "v"(ga) : "memory");
        }
        asm volatile("s_wait_asynccnt 0x0" ::: "memory");
    }
    __syncthreads();

    // ---- GEMM main loop: each wave owns 3 of the 6 column tiles ----
    v8f acc[3];
#pragma unroll
    for (int tt = 0; tt < 3; ++tt)
#pragma unroll
        for (int r = 0; r < 8; ++r) acc[tt][r] = 0.f;

    const int row   = lane & 15;               // A row (n_local)
    const int khalf = (lane >> 4) << 4;        // A lane K offset: 0 or 16

    for (int kc = 0; kc < KCH; ++kc) {
        // A fragment: lanes 0-15 hold K {0..15}, lanes 16-31 hold K {16..31}
        const unsigned short* ap = As + row * K_TOT + kc * 32 + khalf;
        BFrag a;
        a.u[0] = *(const uint4*)(ap);          // K_local +0..7
        a.u[1] = *(const uint4*)(ap + 8);      // K_local +8..15
#pragma unroll
        for (int tt = 0; tt < 3; ++tt) {
            int t = wave * 3 + tt;
            const unsigned short* bp =
                bexp + ((((size_t)g * KCH + kc) * NT + t) * 32 + lane) * 16;
            BFrag b;
            b.u[0] = *(const uint4*)(bp);
            b.u[1] = *(const uint4*)(bp + 8);
            acc[tt] = __builtin_amdgcn_wmma_f32_16x16x32_bf16(
                false, a.v, false, b.v, (short)0, acc[tt], false, false);
        }
    }

    // ---- spill accumulators: C/D layout — VGPR r, lanes 0-15: M=r, lanes 16-31: M=r+8 ----
    const int colw = lane & 15;
    const int mh   = (lane >> 4) * 8;
#pragma unroll
    for (int tt = 0; tt < 3; ++tt) {
        int t = wave * 3 + tt;
#pragma unroll
        for (int r = 0; r < 8; ++r)
            Ls[mh + r][t * 16 + colw] = acc[tt][r];
    }
    __syncthreads();

    // ---- SiLU + mean over 12 positions: 128 (n,oc) tasks over 64 threads ----
    for (int task = tid; task < 16 * OCPG; task += 64) {
        int nl = task >> 3, oc = task & 7;
        float bias = conv_b[g * OCPG + oc];
        float sum = 0.f;
#pragma unroll
        for (int p = 0; p < LONGD; ++p) {
            float y = Ls[nl][p * OCPG + oc] + bias;
            sum += y / (1.f + __expf(-y));     // SiLU
        }
        Pm[nl][oc] = sum * (1.f / (float)LONGD);
    }
    __syncthreads();

    // ---- GroupNorm over the group's 8 channels, per n-row ----
    if (tid < 16) {
        int nl = tid;
        float mu = 0.f;
#pragma unroll
        for (int oc = 0; oc < OCPG; ++oc) mu += Pm[nl][oc];
        mu *= (1.f / (float)OCPG);
        float var = 0.f;
#pragma unroll
        for (int oc = 0; oc < OCPG; ++oc) {
            float d = Pm[nl][oc] - mu;
            var += d * d;
        }
        var *= (1.f / (float)OCPG);
        float rs = rsqrtf(var + EPSF);
        float* o = out + (size_t)(n0 + nl) * C_MID + g * OCPG;
#pragma unroll
        for (int oc = 0; oc < OCPG; ++oc) {
            int ch = g * OCPG + oc;
            o[oc] = (Pm[nl][oc] - mu) * rs * gn_w[ch] + gn_b[ch];
        }
    }
}

// ---------------- host launch ----------------
extern "C" void kernel_launch(void* const* d_in, const int* in_sizes, int n_in,
                              void* d_out, int out_size, void* d_ws, size_t ws_size,
                              hipStream_t stream) {
    const float* x       = (const float*)d_in[0];
    // d_in[1] = out_size scalar (device) — use host out_size instead
    const int*   pidx    = (const int*)d_in[2];
    const float* pdelta  = (const float*)d_in[3];
    const int*   psrc    = (const int*)d_in[4];
    const float* conv_w  = (const float*)d_in[5];
    const float* conv_b  = (const float*)d_in[6];
    const float* gn_w    = (const float*)d_in[7];
    const float* gn_b    = (const float*)d_in[8];

    const int E  = in_sizes[0] / C_IN;
    const int Nn = out_size / C_MID;     // number of targets (4096)

    unsigned short* xproj = (unsigned short*)d_ws;          // bf16 scatter buffer
    size_t xproj_elems = (size_t)Nn * SPATIAL * C_IN;
    size_t xproj_bytes = (xproj_elems * sizeof(unsigned short) + 255) & ~(size_t)255;
    unsigned short* bexp = (unsigned short*)((char*)d_ws + xproj_bytes);

    // 1) zero the scatter buffer (grid-stride 16B)
    long n16 = (long)(xproj_elems * sizeof(unsigned short) / 16);
    zero_ws_kernel<<<4096, 256, 0, stream>>>((uint4*)xproj, n16);

    // 2) expand circular conv weights into pre-swizzled WMMA B fragments (L2-resident, 4.7 MB)
    const int btotal = GROUPSD * KCH * NT * 32 * 16;
    build_bexp_kernel<<<(btotal + 255) / 256, 256, 0, stream>>>(conv_w, bexp);

    // 3) weighted scatter-add (one wave per edge, packed-bf16 global atomics)
    long sthreads = (long)E * 32;
    int  sblocks  = (int)((sthreads + 255) / 256);
    scatter_add_kernel<<<sblocks, 256, 0, stream>>>(x, pidx, pdelta, psrc, xproj, E);

    // 4) fused WMMA conv + SiLU + pool + GroupNorm; g fastest for L2 line sharing
    dim3 grid(GROUPSD, Nn / 16);
    conv_fused_kernel<<<grid, 64, 0, stream>>>(xproj, bexp, conv_b, gn_w, gn_b, (float*)d_out);
}
